// SpatialAttention_16166256902486
// MI455X (gfx1250) — matrix operand order repaired and checked by reading, not compiled, not run on hardware
//
#include <hip/hip_runtime.h>
#include <hip/hip_bf16.h>

#define Bc    64
#define Cc    32
#define Nc    512
#define Tc    12
#define COUTc 32
#define ALPHAc 0.05f

typedef _Float16 h8   __attribute__((ext_vector_type(8)));
typedef _Float16 v16h __attribute__((ext_vector_type(16)));
typedef float    v8f  __attribute__((ext_vector_type(8)));

// ---------------------------------------------------------------------------
// K0: relu (float4 vectorized)
// ---------------------------------------------------------------------------
__global__ __launch_bounds__(256) void k_relu(const float* __restrict__ x,
                                              float* __restrict__ xr, int n4) {
  int i = blockIdx.x * 256 + threadIdx.x;
  if (i < n4) {
    float4 v = ((const float4*)x)[i];
    v.x = fmaxf(v.x, 0.f); v.y = fmaxf(v.y, 0.f);
    v.z = fmaxf(v.z, 0.f); v.w = fmaxf(v.w, 0.f);
    ((float4*)xr)[i] = v;
  }
}

// ---------------------------------------------------------------------------
// K1: lhs[b,n,t] = sum_c (sum_t' xr[b,c,n,t']*W1[t']) * W2[c,t]
//     rhs[b,n,t] = sum_c W3[c]*xr[b,c,n,t]        (one wave per (b,n))
// ---------------------------------------------------------------------------
__global__ __launch_bounds__(32) void k_lhs_rhs(const float* __restrict__ xr,
                                                const float* __restrict__ W1,
                                                const float* __restrict__ W2,
                                                const float* __restrict__ W3,
                                                float* __restrict__ lhs,
                                                float* __restrict__ rhs) {
  int bn = blockIdx.x;            // b*Nc + n
  int b  = bn >> 9, n = bn & (Nc - 1);
  int c  = threadIdx.x;           // 0..31
  __shared__ float rowbuf[Cc][Tc];
  __shared__ float tmpbuf[Cc];
  const float* src = xr + (((size_t)b * Cc + c) * Nc + n) * Tc;
  float w3 = W3[c], tmp = 0.f;
#pragma unroll
  for (int t = 0; t < Tc; ++t) {
    float v = src[t];
    rowbuf[c][t] = v * w3;
    tmp += v * W1[t];
  }
  tmpbuf[c] = tmp;
  __syncthreads();
  if (c < Tc) {
    float l = 0.f, r = 0.f;
#pragma unroll
    for (int cc = 0; cc < Cc; ++cc) {
      l += tmpbuf[cc] * W2[cc * Tc + c];
      r += rowbuf[cc][c];
    }
    lhs[(size_t)bn * Tc + c] = l;
    rhs[(size_t)bn * Tc + c] = r;
  }
}

// ---------------------------------------------------------------------------
// K2: product[b,i,j] = sum_t lhs[b,i,t]*rhs[b,j,t];
//     SIGT16[b,j,i] = f16(sigmoid(product + bs[i,j]))   (transposed store!)
// ---------------------------------------------------------------------------
__global__ __launch_bounds__(256) void k_prod_sig(const float* __restrict__ lhs,
                                                  const float* __restrict__ rhs,
                                                  const float* __restrict__ bs,
                                                  _Float16* __restrict__ SIGT) {
  __shared__ float ls[16][Tc], rs[16][Tc];
  int b = blockIdx.z, i0 = blockIdx.y * 16, j0 = blockIdx.x * 16;
  int tid = threadIdx.x;
  if (tid < 192) {
    int r = tid / Tc, t = tid - r * Tc;
    ls[r][t] = lhs[((size_t)b * Nc + i0 + r) * Tc + t];
    rs[r][t] = rhs[((size_t)b * Nc + j0 + r) * Tc + t];
  }
  __syncthreads();
  int ti = tid & 15, tj = tid >> 4;
  float p = 0.f;
#pragma unroll
  for (int t = 0; t < Tc; ++t) p += ls[ti][t] * rs[tj][t];
  p += bs[(size_t)(i0 + ti) * Nc + (j0 + tj)];
  float s = 1.f / (1.f + __expf(-p));
  SIGT[(size_t)b * Nc * Nc + (size_t)(j0 + tj) * Nc + (i0 + ti)] = (_Float16)s;
}

// ---------------------------------------------------------------------------
// K3: f32 -> f16 elementwise
// ---------------------------------------------------------------------------
__global__ __launch_bounds__(256) void k_conv16(const float* __restrict__ in,
                                                _Float16* __restrict__ out, int n) {
  int i = blockIdx.x * 256 + threadIdx.x;
  if (i < n) out[i] = (_Float16)in[i];
}

// ---------------------------------------------------------------------------
// K4: generic WMMA NT-GEMM, f16 inputs, f32 accumulate.
//   D[b, m, n] = sum_k A[b, m, k] * Bt[b, n, k]        K = Nc = 512
//   Block tile 64x64, 8 waves, each wave 2 adjacent N-tiles (shared A-frag).
//   mode 0: D[b*sD + m*Nc + n] = acc
//   mode 1: mixprop epilogue: n -> (c = n/Tc, t = n%Tc);
//           off = b*sD + (c*Nc + m)*Tc + t;  D[off] = a*xr[off] + (1-a)*acc
// ---------------------------------------------------------------------------
__global__ __launch_bounds__(256)
void k_wmma_gemm(const _Float16* __restrict__ A,  size_t sA,
                 const _Float16* __restrict__ Bt, size_t sB,
                 float* __restrict__ D,           size_t sD,
                 int mode, const float* __restrict__ xr) {
  __shared__ _Float16 As[64][40];   // +8 halves pad, rows stay 16B-aligned
  __shared__ _Float16 Bs[64][40];

  const int tid = threadIdx.x;
  const int bz  = blockIdx.z;
  const int m0  = blockIdx.y * 64;
  const int n0  = blockIdx.x * 64;
  A  += (size_t)bz * sA;
  Bt += (size_t)bz * sB;

  const int wave = tid >> 5, lane = tid & 31;
  const int mi    = wave >> 1;          // 0..3 : M sub-tile
  const int njb   = (wave & 1) * 2;     // 0 or 2 : first of two N sub-tiles
  const int rowA  = mi * 16 + (lane & 15);
  const int koffA = (lane >> 4) * 16;   // A frag: 16 contiguous halves at koff
  const int colB0 = njb * 16 + (lane & 15);
  const int colB1 = colB0 + 16;
  const int kbB   = (lane >> 4) * 16;   // B frag: 16 contiguous halves at kbase

  const int ldrow = tid >> 2;           // 0..63, 8 halves per thread per tile
  const int ldseg = (tid & 3) * 8;

  v8f acc0 = {}; v8f acc1 = {};

  for (int kk = 0; kk < Nc; kk += 32) {
    *(h8*)&As[ldrow][ldseg] = *(const h8*)(A  + (size_t)(m0 + ldrow) * Nc + kk + ldseg);
    *(h8*)&Bs[ldrow][ldseg] = *(const h8*)(Bt + (size_t)(n0 + ldrow) * Nc + kk + ldseg);
    if (kk + 32 < Nc) {
      __builtin_prefetch(A  + (size_t)(m0 + ldrow) * Nc + kk + 32 + ldseg, 0, 1);
      __builtin_prefetch(Bt + (size_t)(n0 + ldrow) * Nc + kk + 32 + ldseg, 0, 1);
    }
    __syncthreads();

    h8 alo = *(const h8*)&As[rowA][koffA];
    h8 ahi = *(const h8*)&As[rowA][koffA + 8];
    v16h a = __builtin_shufflevector(alo, ahi, 0,1,2,3,4,5,6,7,8,9,10,11,12,13,14,15);

    h8 b0lo = *(const h8*)&Bs[colB0][kbB];
    h8 b0hi = *(const h8*)&Bs[colB0][kbB + 8];
    v16h bf0 = __builtin_shufflevector(b0lo, b0hi, 0,1,2,3,4,5,6,7,8,9,10,11,12,13,14,15);

    h8 b1lo = *(const h8*)&Bs[colB1][kbB];
    h8 b1hi = *(const h8*)&Bs[colB1][kbB + 8];
    v16h bf1 = __builtin_shufflevector(b1lo, b1hi, 0,1,2,3,4,5,6,7,8,9,10,11,12,13,14,15);

    acc0 = __builtin_amdgcn_wmma_f32_16x16x32_f16(false, a, false, bf0, (short)0, acc0, false, false);
    acc1 = __builtin_amdgcn_wmma_f32_16x16x32_f16(false, a, false, bf1, (short)0, acc1, false, false);
    __syncthreads();
  }

  // C/D layout: VGPR r -> row = r + 8*(lane>>4); col = lane&15 (per N sub-tile)
  const int crow  = m0 + mi * 16 + 8 * (lane >> 4);
  const int ccol0 = n0 + njb * 16 + (lane & 15);
  const int ccol1 = ccol0 + 16;

  if (mode == 0) {
    float* Db = D + (size_t)bz * sD;
#pragma unroll
    for (int r = 0; r < 8; ++r) {
      Db[(size_t)(crow + r) * Nc + ccol0] = acc0[r];
      Db[(size_t)(crow + r) * Nc + ccol1] = acc1[r];
    }
  } else {
    float*       Db = D  + (size_t)bz * sD;
    const float* Xb = xr + (size_t)bz * sD;
    const int c0 = ccol0 / Tc, t0 = ccol0 - c0 * Tc;
    const int c1 = ccol1 / Tc, t1 = ccol1 - c1 * Tc;
#pragma unroll
    for (int r = 0; r < 8; ++r) {
      size_t o0 = ((size_t)c0 * Nc + crow + r) * Tc + t0;
      size_t o1 = ((size_t)c1 * Nc + crow + r) * Tc + t1;
      Db[o0] = ALPHAc * Xb[o0] + (1.f - ALPHAc) * acc0[r];
      Db[o1] = ALPHAc * Xb[o1] + (1.f - ALPHAc) * acc1[r];
    }
  }
}

// ---------------------------------------------------------------------------
// K5: in-place softmax over first N index (axis=1): S[b,:,k]  (online)
// ---------------------------------------------------------------------------
__global__ __launch_bounds__(256) void k_colsoftmax(float* __restrict__ S) {
  int b = blockIdx.y;
  int k = blockIdx.x * 256 + threadIdx.x;
  float* col = S + (size_t)b * Nc * Nc + k;
  float m = -1e30f, d = 0.f;
  for (int n = 0; n < Nc; ++n) {
    float v  = col[(size_t)n * Nc];
    float nm = fmaxf(m, v);
    d = d * __expf(m - nm) + __expf(v - nm);
    m = nm;
  }
  float inv = 1.f / d;
  for (int n = 0; n < Nc; ++n) {
    float v = col[(size_t)n * Nc];
    col[(size_t)n * Nc] = __expf(v - m) * inv;
  }
}

// ---------------------------------------------------------------------------
// K6 (branch 1): AT16[b,w,v] = f16(S_norm[b,v,w])  — tiled transpose-convert
// ---------------------------------------------------------------------------
__global__ __launch_bounds__(256) void k_transconv(const float* __restrict__ S,
                                                   _Float16* __restrict__ AT) {
  __shared__ float tile[32][33];
  int b  = blockIdx.z;
  int w0 = blockIdx.x * 32, v0 = blockIdx.y * 32;
  int tx = threadIdx.x, ty = threadIdx.y;   // 32 x 8
  const float* Sb = S + (size_t)b * Nc * Nc;
#pragma unroll
  for (int r = 0; r < 32; r += 8)
    tile[ty + r][tx] = Sb[(size_t)(v0 + ty + r) * Nc + (w0 + tx)];
  __syncthreads();
  _Float16* Ab = AT + (size_t)b * Nc * Nc;
#pragma unroll
  for (int r = 0; r < 32; r += 8)
    Ab[(size_t)(w0 + ty + r) * Nc + (v0 + tx)] = (_Float16)tile[tx][ty + r];
}

// ---------------------------------------------------------------------------
// K7 (branch 2): AT16[b,w,v] = softmax over w of S_norm[b,w,v], written f16.
//   (adj1 = softmax(S_norm^T, axis=2); stored already transposed for GEMM.)
// ---------------------------------------------------------------------------
__global__ __launch_bounds__(256) void k_coladj(const float* __restrict__ S,
                                                _Float16* __restrict__ AT) {
  int b = blockIdx.y;
  int v = blockIdx.x * 256 + threadIdx.x;
  const float* col = S + (size_t)b * Nc * Nc + v;
  _Float16*    oc  = AT + (size_t)b * Nc * Nc + v;
  float m = -1e30f, d = 0.f;
  for (int w = 0; w < Nc; ++w) {
    float x  = col[(size_t)w * Nc];
    float nm = fmaxf(m, x);
    d = d * __expf(m - nm) + __expf(x - nm);
    m = nm;
  }
  float inv = 1.f / d;
  for (int w = 0; w < Nc; ++w) {
    float x = col[(size_t)w * Nc];
    oc[(size_t)w * Nc] = (_Float16)(__expf(x - m) * inv);
  }
}

// ---------------------------------------------------------------------------
// K8: H16[b, c*Tc+t, v] = f16(h[b,c,v,t])   (make K(=v) contiguous for WMMA B)
// ---------------------------------------------------------------------------
__global__ __launch_bounds__(256) void k_convh(const float* __restrict__ h,
                                               _Float16* __restrict__ H16) {
  int b = blockIdx.z, c = blockIdx.y;
  int v = blockIdx.x * 256 + threadIdx.x;
  const float4* src = (const float4*)(h + (((size_t)b * Cc + c) * Nc + v) * Tc);
  float4 f0 = src[0], f1 = src[1], f2 = src[2];
  float vals[Tc] = {f0.x, f0.y, f0.z, f0.w, f1.x, f1.y, f1.z, f1.w, f2.x, f2.y, f2.z, f2.w};
  _Float16* dst = H16 + ((size_t)b * Cc + c) * Tc * Nc + v;
#pragma unroll
  for (int t = 0; t < Tc; ++t) dst[(size_t)t * Nc] = (_Float16)vals[t];
}

// ---------------------------------------------------------------------------
// K9: out[b,o,n,t] += sum_c W[o, colblk*Cc + c] * h[b,c,n,t]   (LDS tiled)
// ---------------------------------------------------------------------------
__global__ __launch_bounds__(256) void k_mlp_accum(const float* __restrict__ h,
                                                   const float* __restrict__ W,
                                                   int colblk,
                                                   float* __restrict__ out) {
  __shared__ float hs[Cc][16 * Tc];   // 24 KB
  __shared__ float wsm[COUTc][Cc];
  int b = blockIdx.y, n0 = blockIdx.x * 16;
  int tid = threadIdx.x;
  for (int idx = tid; idx < Cc * 16 * Tc; idx += 256) {
    int c = idx / (16 * Tc), rem = idx - c * (16 * Tc);
    hs[c][rem] = h[(((size_t)b * Cc + c) * Nc + n0) * Tc + rem];
  }
  for (int idx = tid; idx < COUTc * Cc; idx += 256) {
    int o = idx >> 5, c = idx & 31;
    wsm[o][c] = W[o * (4 * Cc) + colblk * Cc + c];
  }
  __syncthreads();
  for (int idx = tid; idx < COUTc * 16 * Tc; idx += 256) {
    int o = idx / (16 * Tc), rem = idx - o * (16 * Tc);
    float acc = 0.f;
#pragma unroll
    for (int c = 0; c < Cc; ++c) acc += wsm[o][c] * hs[c][rem];
    size_t ooff = (((size_t)b * COUTc + o) * Nc + n0) * Tc + rem;
    out[ooff] += acc;
  }
}

// ---------------------------------------------------------------------------
// K_init: out = mlp1_b[o] + mlp2_b[o]  (both branch biases)
// ---------------------------------------------------------------------------
__global__ __launch_bounds__(256) void k_init_out(const float* __restrict__ b1,
                                                  const float* __restrict__ b2,
                                                  float* __restrict__ out, int total) {
  int i = blockIdx.x * 256 + threadIdx.x;
  if (i < total) {
    int o = (i / (Nc * Tc)) % COUTc;
    out[i] = b1[o] + b2[o];
  }
}

// ---------------------------------------------------------------------------
extern "C" void kernel_launch(void* const* d_in, const int* in_sizes, int n_in,
                              void* d_out, int out_size, void* d_ws, size_t ws_size,
                              hipStream_t stream) {
  (void)in_sizes; (void)n_in; (void)ws_size;
  const float* x   = (const float*)d_in[0];
  const float* W1  = (const float*)d_in[1];
  const float* W2  = (const float*)d_in[2];
  const float* W3  = (const float*)d_in[3];
  const float* bs  = (const float*)d_in[4];
  const float* Vs  = (const float*)d_in[5];
  const float* m1w = (const float*)d_in[6];
  const float* m1b = (const float*)d_in[7];
  const float* m2w = (const float*)d_in[8];
  const float* m2b = (const float*)d_in[9];
  float* out = (float*)d_out;

  // ---- workspace carve (≈290 MB) ----
  char* wp = (char*)d_ws;
  size_t off = 0;
  auto carve = [&](size_t bytes) -> char* {
    off = (off + 255) & ~(size_t)255;
    char* p = wp + off;
    off += bytes;
    return p;
  };
  const size_t SZ_X  = (size_t)Bc * Cc * Nc * Tc;   // 12,582,912 elems
  const size_t SZ_NN = (size_t)Bc * Nc * Nc;        // 16,777,216 elems
  float*    XR     = (float*)carve(SZ_X * 4);
  float*    Hp     = (float*)carve(SZ_X * 4);
  float*    Hq     = (float*)carve(SZ_X * 4);
  float*    S      = (float*)carve(SZ_NN * 4);
  _Float16* AT16   = (_Float16*)carve(SZ_NN * 2);
  _Float16* SIGT16 = (_Float16*)carve(SZ_NN * 2);
  _Float16* H16    = SIGT16;                         // reuse: SIGT dead after GEMM-S
  _Float16* VS16   = (_Float16*)carve((size_t)Nc * Nc * 2);
  float*    LHS    = (float*)carve((size_t)Bc * Nc * Tc * 4);
  float*    RHS    = (float*)carve((size_t)Bc * Nc * Tc * 4);

  // 1. relu
  k_relu<<<(int)((SZ_X / 4 + 255) / 256), 256, 0, stream>>>(x, XR, (int)(SZ_X / 4));
  // 2. lhs/rhs projections
  k_lhs_rhs<<<Bc * Nc, 32, 0, stream>>>(XR, W1, W2, W3, LHS, RHS);
  // 3. product + sigmoid + bias, transposed f16 store
  k_prod_sig<<<dim3(Nc / 16, Nc / 16, Bc), 256, 0, stream>>>(LHS, RHS, bs, SIGT16);
  // 4. Vs -> f16
  k_conv16<<<(Nc * Nc + 255) / 256, 256, 0, stream>>>(Vs, VS16, Nc * Nc);
  // 5. S = Vs @ sig  (WMMA, mode 0)
  k_wmma_gemm<<<dim3(Nc / 64, Nc / 64, Bc), 256, 0, stream>>>(
      VS16, (size_t)0, SIGT16, (size_t)Nc * Nc, S, (size_t)Nc * Nc, 0, nullptr);
  // 6. softmax over axis=1 (in place -> S_norm)
  k_colsoftmax<<<dim3(Nc / 256, Bc), 256, 0, stream>>>(S);
  // 7. out = bias1 + bias2
  k_init_out<<<(out_size + 255) / 256, 256, 0, stream>>>(m1b, m2b, out, out_size);

  // 8. two mixprop branches
  for (int br = 0; br < 2; ++br) {
    const float* mw = (br == 0) ? m1w : m2w;
    if (br == 0)
      k_transconv<<<dim3(Nc / 32, Nc / 32, Bc), dim3(32, 8), 0, stream>>>(S, AT16);
    else
      k_coladj<<<dim3(Nc / 256, Bc), 256, 0, stream>>>(S, AT16);

    k_mlp_accum<<<dim3(Nc / 16, Bc), 256, 0, stream>>>(XR, mw, 0, out);

    const float* hprev = XR;
    for (int k = 1; k <= 3; ++k) {
      float* hcur = (k & 1) ? Hp : Hq;
      k_convh<<<dim3(Nc / 256, Cc, Bc), 256, 0, stream>>>(hprev, H16);
      // h_next = alpha*x + (1-alpha) * (A^T-major WMMA GEMM), fused epilogue
      k_wmma_gemm<<<dim3((Cc * Tc) / 64, Nc / 64, Bc), 256, 0, stream>>>(
          AT16, (size_t)Nc * Nc, H16, (size_t)Cc * Tc * Nc,
          hcur, (size_t)Cc * Nc * Tc, 1, XR);
      k_mlp_accum<<<dim3(Nc / 16, Bc), 256, 0, stream>>>(hcur, mw, k, out);
      hprev = hcur;
    }
  }
}
